// SphericalGlobalAttention_77283641524743
// MI455X (gfx1250) — matrix operand order repaired and checked by reading, not compiled, not run on hardware
//
#include <hip/hip_runtime.h>
#include <hip/hip_bf16.h>
#include <math.h>

typedef __attribute__((ext_vector_type(16))) _Float16 v16h;
typedef __attribute__((ext_vector_type(8)))  _Float16 v8h;
typedef __attribute__((ext_vector_type(8)))  float    v8f;

#define DIMC 768
#define NH   12
#define HD   64
#define BB   4
#define NN   2048
#define BN   (BB*NN)    /* 8192 rows of x */
#define C3   (3*DIMC)   /* 2304 */
#define ATT_SCALE 0.125f

// ---------- fragment loaders (wave32 WMMA, CDNA5 ISA; k order is this code's own, same for A and B) ----------
// A (16-bit 16x32): lane L: row = L&15; K = h*16+(0..15) contiguous, h = L>>4.
static __device__ __forceinline__ v16h fragA(const _Float16* p) {
  v8h lo = *(const v8h*)(p);
  v8h hi = *(const v8h*)(p + 8);
  return __builtin_shufflevector(lo, hi, 0,1,2,3,4,5,6,7,8,9,10,11,12,13,14,15);
}
// B (16-bit 32x16): lane L: col = L&15; K = h*16+(0..15) contiguous, h = L>>4.
static __device__ __forceinline__ v16h fragB(const _Float16* p) {
  v8h lo = *(const v8h*)(p);
  v8h hi = *(const v8h*)(p + 8);
  return __builtin_shufflevector(lo, hi, 0,1,2,3,4,5,6,7,8,9,10,11,12,13,14,15);
}
static __device__ __forceinline__ v8f wmma16(v16h a, v16h b, v8f c) {
  return __builtin_amdgcn_wmma_f32_16x16x32_f16(false, a, false, b, (short)0, c, false, false);
}

// ---------- CDNA5 async global->LDS copy (GLOBAL_LOAD_ASYNC_TO_LDS_B128, ASYNCcnt) ----
static __device__ __forceinline__ void async_b128(const _Float16* g, const _Float16* l) {
  unsigned lds = (unsigned)(unsigned long long)l;          // low 32 bits = LDS offset
  unsigned long long ga = (unsigned long long)g;
  asm volatile("global_load_async_to_lds_b128 %0, %1, off"
               :: "v"(lds), "v"(ga) : "memory");
}
static __device__ __forceinline__ void wait_async0() {
  asm volatile("s_wait_asynccnt 0x0" ::: "memory");
}

// ---------- prep kernels ----------
__global__ void __launch_bounds__(256) cvt_f16_kernel(const float* __restrict__ src,
                                                      _Float16* __restrict__ dst, long n) {
  long i = (long)blockIdx.x * 256 + threadIdx.x;
  if (i < n) dst[i] = (_Float16)src[i];
}
// dst[n*K + k] = (f16) src[k*Ncols + n]   (weights -> [n][k] f16)
__global__ void __launch_bounds__(256) transpose_f16_kernel(const float* __restrict__ src,
                                                            _Float16* __restrict__ dst,
                                                            int K, int Ncols) {
  long i = (long)blockIdx.x * 256 + threadIdx.x;
  long total = (long)K * Ncols;
  if (i < total) {
    long n = i / K, k = i % K;
    dst[i] = (_Float16)src[k * (long)Ncols + n];
  }
}
__global__ void __launch_bounds__(256) logw_kernel(const float* __restrict__ w,
                                                   float* __restrict__ lw, int n) {
  int i = blockIdx.x * 256 + threadIdx.x;
  if (i < n) lw[i] = __logf(w[i]);
}

// ---------- shared GEMM mainloop: 256 thr = 8 waves, tile 128x64, K-step 32 ----------
// Double-buffered LDS, async global->LDS copies overlapped with WMMA.
static __device__ __forceinline__ void gemm_mainloop(
    const _Float16* __restrict__ Ag,   // [M][K] row-major f16
    const _Float16* __restrict__ BTg,  // [N][K] (B transposed) f16
    int K, long rowBase, long colBase,
    _Float16* ldsA /*2*128*32*/, _Float16* ldsB /*2*64*32*/, v8f acc[4])
{
  const int t = threadIdx.x;
  const int w = t >> 5, L = t & 31, half = L >> 4, l15 = L & 15;
  for (int g = 0; g < 4; ++g) for (int i = 0; i < 8; ++i) acc[g][i] = 0.0f;

  const int arow = t >> 1, akp = (t & 1) * 16;   // A tile: 128x32, 32B/thread
  const int bn   = t >> 2, bkp = (t & 3) * 8;    // B tile: 64x32, 16B/thread
  const _Float16* aSrc = Ag  + (rowBase + arow) * (long)K + akp;
  const _Float16* bSrc = BTg + (colBase + bn)   * (long)K + bkp;
  _Float16* laBase = ldsA + arow * 32 + akp;
  _Float16* lbBase = ldsB + bn * 32 + bkp;

  auto issue = [&](int buf, int k0) {
    async_b128(aSrc + k0,     laBase + buf * (128 * 32));
    async_b128(aSrc + k0 + 8, laBase + buf * (128 * 32) + 8);
    async_b128(bSrc + k0,     lbBase + buf * (64 * 32));
  };

  issue(0, 0);
  int cur = 0;
  for (int k0 = 0; k0 < K; k0 += 32) {
    wait_async0();          // own async copies for buffer `cur` landed in LDS
    __syncthreads();        // all waves' copies landed; all done reading other buffer
    if (k0 + 32 < K) issue(cur ^ 1, k0 + 32);
    const _Float16* lA = ldsA + cur * (128 * 32);
    const _Float16* lB = ldsB + cur * (64 * 32);
    v16h a = fragA(lA + (w * 16 + l15) * 32 + half * 16);
    for (int g = 0; g < 4; ++g) {
      v16h b = fragB(lB + (g * 16 + l15) * 32 + half * 16);
      acc[g] = wmma16(a, b, acc[g]);
    }
    cur ^= 1;
  }
}

// ---------- QKV GEMM: [8192,768] @ [768,2304] -> scatter Q,K [B,H,N,D], V^T [B,H,D,N]
__global__ void __launch_bounds__(256) qkv_gemm_kernel(
    const _Float16* __restrict__ Xh, const _Float16* __restrict__ WqkvT,
    _Float16* __restrict__ Qh, _Float16* __restrict__ Kh, _Float16* __restrict__ Vt)
{
  __shared__ _Float16 ldsA[2 * 128 * 32];
  __shared__ _Float16 ldsB[2 * 64 * 32];
  v8f acc[4];
  long rowBase = (long)blockIdx.x * 128;
  long colBase = (long)blockIdx.y * 64;
  gemm_mainloop(Xh, WqkvT, DIMC, rowBase, colBase, ldsA, ldsB, acc);

  const int t = threadIdx.x, w = t >> 5, L = t & 31, half = L >> 4, l15 = L & 15;
  for (int g = 0; g < 4; ++g) {
    long gc = colBase + g * 16 + l15;        // 0..2303
    int s = (int)(gc / DIMC);
    int rem = (int)(gc % DIMC);
    int h = rem >> 6, d = rem & 63;
    for (int r = 0; r < 8; ++r) {
      long gr = rowBase + w * 16 + r + 8 * half;   // 0..8191
      int b = (int)(gr >> 11), n = (int)(gr & 2047);
      _Float16 val = (_Float16)acc[g][r];
      long hb = (long)(b * NH + h);
      if (s == 0)      Qh[(hb * NN + n) * HD + d] = val;
      else if (s == 1) Kh[(hb * NN + n) * HD + d] = val;
      else             Vt[(hb * HD + d) * NN + n] = val;
    }
  }
}

// ---------- flash attention: block = (b,h) x 128 query rows, K/V chunks of 32 ----------
__global__ void __launch_bounds__(256) attn_kernel(
    const _Float16* __restrict__ Qh, const _Float16* __restrict__ Kh,
    const _Float16* __restrict__ Vt, const float* __restrict__ logw,
    _Float16* __restrict__ Ah)
{
  __shared__ _Float16 Qs[128 * 64];          // 16 KB
  __shared__ _Float16 Ks[2 * 32 * 64];       // 8 KB double-buffered K chunk
  __shared__ _Float16 Vs[2 * 64 * 32];       // 8 KB double-buffered V^T chunk [d][m]
  __shared__ _Float16 Ps[8 * 16 * 32];       // 8 KB wave-private P tiles

  const int bh = blockIdx.x;                 // 0..47
  const int b = bh / NH, head = bh % NH;
  const long nbase = (long)blockIdx.y * 128;
  const int t = threadIdx.x, w = t >> 5, L = t & 31, half = L >> 4, l15 = L & 15;

  const _Float16* Qg = Qh + ((long)bh * NN + nbase) * HD;
  const _Float16* Kg = Kh + (long)bh * NN * HD;
  const _Float16* Vg = Vt + (long)bh * HD * NN;

  const int krow = t >> 3, kdp = (t & 7) * 8;   // K chunk 32x64: 16B/thread
  const int vd   = t >> 2, vjp = (t & 3) * 8;   // V chunk 64x32: 16B/thread
  auto issue_kv = [&](int buf, int mb) {
    async_b128(Kg + (long)(mb + krow) * HD + kdp, Ks + buf * 2048 + krow * 64 + kdp);
    async_b128(Vg + (long)vd * NN + mb + vjp,     Vs + buf * 2048 + vd * 32 + vjp);
  };

  { // load Q tile 128x64 (each thread: 32 f16)
    int row = t >> 1, dp = (t & 1) * 32;
    const _Float16* src = Qg + row * HD + dp;
    for (int j = 0; j < 4; ++j)
      *(v8h*)(Qs + row * 64 + dp + j * 8) = *(const v8h*)(src + j * 8);
  }
  issue_kv(0, 0);
  __syncthreads();
  v16h qa0 = fragA(Qs + (w * 16 + l15) * 64 + 0 * 32 + half * 16);
  v16h qa1 = fragA(Qs + (w * 16 + l15) * 64 + 1 * 32 + half * 16);

  float mrow[8], lrow[8];
  v8f o[4];
  for (int r = 0; r < 8; ++r) { mrow[r] = -1e30f; lrow[r] = 0.0f; }
  for (int g = 0; g < 4; ++g) for (int i = 0; i < 8; ++i) o[g][i] = 0.0f;

  int cur = 0;
  for (int mb = 0; mb < NN; mb += 32) {
    wait_async0();
    __syncthreads();
    if (mb + 32 < NN) issue_kv(cur ^ 1, mb + 32);
    const _Float16* Kc = Ks + cur * 2048;
    const _Float16* Vc = Vs + cur * 2048;

    // S = Q * K^T  (16x32 per wave, as two 16x16 f32 frags)
    v8f s0, s1;
    for (int i = 0; i < 8; ++i) { s0[i] = 0.0f; s1[i] = 0.0f; }
    {
      v16h b00 = fragB(Kc + (l15)      * 64 + 0 * 32 + half * 16);
      v16h b01 = fragB(Kc + (l15)      * 64 + 1 * 32 + half * 16);
      v16h b10 = fragB(Kc + (16 + l15) * 64 + 0 * 32 + half * 16);
      v16h b11 = fragB(Kc + (16 + l15) * 64 + 1 * 32 + half * 16);
      s0 = wmma16(qa0, b00, s0); s0 = wmma16(qa1, b01, s0);
      s1 = wmma16(qa0, b10, s1); s1 = wmma16(qa1, b11, s1);
    }

    // scale + log-quadrature bias (per key column)
    float bias0 = logw[mb + l15];
    float bias1 = logw[mb + 16 + l15];
    float p0[8], p1[8], mx[8], rs[8], mnew[8], alpha[8];
    for (int r = 0; r < 8; ++r) {
      s0[r] = s0[r] * ATT_SCALE + bias0;
      s1[r] = s1[r] * ATT_SCALE + bias1;
      mx[r] = fmaxf(s0[r], s1[r]);
    }
    // row-wise reductions stay inside the 16-lane half owning each row group
    for (int mask = 1; mask < 16; mask <<= 1)
      for (int r = 0; r < 8; ++r) mx[r] = fmaxf(mx[r], __shfl_xor(mx[r], mask, 32));
    for (int r = 0; r < 8; ++r) {
      mnew[r]  = fmaxf(mrow[r], mx[r]);
      alpha[r] = __expf(mrow[r] - mnew[r]);
      p0[r]    = __expf(s0[r] - mnew[r]);
      p1[r]    = __expf(s1[r] - mnew[r]);
      rs[r]    = p0[r] + p1[r];
    }
    for (int mask = 1; mask < 16; mask <<= 1)
      for (int r = 0; r < 8; ++r) rs[r] += __shfl_xor(rs[r], mask, 32);
    for (int r = 0; r < 8; ++r) {
      lrow[r] = lrow[r] * alpha[r] + rs[r];
      mrow[r] = mnew[r];
    }
    for (int g = 0; g < 4; ++g)
      for (int r = 0; r < 8; ++r) o[g][r] *= alpha[r];

    // P: C-layout -> A-layout via wave-private LDS tile (same-wave DS ordering)
    _Float16* pw = Ps + w * 512;
    for (int r = 0; r < 8; ++r) {
      pw[(r + 8 * half) * 32 + l15]      = (_Float16)p0[r];
      pw[(r + 8 * half) * 32 + 16 + l15] = (_Float16)p1[r];
    }
    v16h pa = fragA(pw + l15 * 32 + half * 16);

    // O += P * V  (K = 32 key positions)
    for (int g2 = 0; g2 < 4; ++g2) {
      v16h vb = fragB(Vc + (g2 * 16 + l15) * 32 + half * 16);
      o[g2] = wmma16(pa, vb, o[g2]);
    }
    cur ^= 1;
  }

  // epilogue: normalize and store f16 attention output [B,N,C]
  for (int g2 = 0; g2 < 4; ++g2) {
    int c = head * 64 + g2 * 16 + l15;
    for (int r = 0; r < 8; ++r) {
      long n = nbase + w * 16 + r + 8 * half;
      float val = o[g2][r] / lrow[r];
      Ah[((long)b * NN + n) * DIMC + c] = (_Float16)val;
    }
  }
}

// ---------- output projection: [8192,768] @ [768,768] + bias -> f32 ----------
__global__ void __launch_bounds__(256) proj_gemm_kernel(
    const _Float16* __restrict__ Ah, const _Float16* __restrict__ WprojT,
    const float* __restrict__ bias, float* __restrict__ out)
{
  __shared__ _Float16 ldsA[2 * 128 * 32];
  __shared__ _Float16 ldsB[2 * 64 * 32];
  v8f acc[4];
  long rowBase = (long)blockIdx.x * 128;
  long colBase = (long)blockIdx.y * 64;
  gemm_mainloop(Ah, WprojT, DIMC, rowBase, colBase, ldsA, ldsB, acc);

  const int t = threadIdx.x, w = t >> 5, L = t & 31, half = L >> 4, l15 = L & 15;
  for (int g = 0; g < 4; ++g) {
    long gc = colBase + g * 16 + l15;
    float bv = bias[gc];
    for (int r = 0; r < 8; ++r) {
      long gr = rowBase + w * 16 + r + 8 * half;
      out[gr * DIMC + gc] = acc[g][r] + bv;
    }
  }
}

// ---------- host ----------
extern "C" void kernel_launch(void* const* d_in, const int* in_sizes, int n_in,
                              void* d_out, int out_size, void* d_ws, size_t ws_size,
                              hipStream_t stream) {
  (void)in_sizes; (void)n_in; (void)out_size; (void)ws_size;
  const float* x     = (const float*)d_in[0];
  const float* qw    = (const float*)d_in[1];
  const float* Wqkv  = (const float*)d_in[2];
  const float* Wproj = (const float*)d_in[3];
  const float* bproj = (const float*)d_in[4];
  float* out = (float*)d_out;

  char* ws = (char*)d_ws;
  size_t off = 0;
  auto wsalloc = [&](size_t bytes) -> char* {
    char* p = ws + off;
    off += (bytes + 255) & ~(size_t)255;
    return p;
  };
  _Float16* Xh     = (_Float16*)wsalloc((size_t)BN * DIMC * 2);
  _Float16* WqkvT  = (_Float16*)wsalloc((size_t)C3 * DIMC * 2);
  _Float16* WprojT = (_Float16*)wsalloc((size_t)DIMC * DIMC * 2);
  float*    lw     = (float*)   wsalloc((size_t)NN * 4);
  size_t qkvElems  = (size_t)BB * NH * NN * HD;
  _Float16* Qh     = (_Float16*)wsalloc(qkvElems * 2);
  _Float16* Kh     = (_Float16*)wsalloc(qkvElems * 2);
  _Float16* Vt     = (_Float16*)wsalloc(qkvElems * 2);
  _Float16* Ah     = (_Float16*)wsalloc((size_t)BN * DIMC * 2);

  { long n = (long)BN * DIMC;
    cvt_f16_kernel<<<dim3((unsigned)((n + 255) / 256)), dim3(256), 0, stream>>>(x, Xh, n); }
  { long n = (long)C3 * DIMC;
    transpose_f16_kernel<<<dim3((unsigned)((n + 255) / 256)), dim3(256), 0, stream>>>(Wqkv, WqkvT, DIMC, C3); }
  { long n = (long)DIMC * DIMC;
    transpose_f16_kernel<<<dim3((unsigned)((n + 255) / 256)), dim3(256), 0, stream>>>(Wproj, WprojT, DIMC, DIMC); }
  logw_kernel<<<dim3(NN / 256), dim3(256), 0, stream>>>(qw, lw, NN);

  qkv_gemm_kernel<<<dim3(BN / 128, C3 / 64), dim3(256), 0, stream>>>(Xh, WqkvT, Qh, Kh, Vt);
  attn_kernel<<<dim3(BB * NH, NN / 128), dim3(256), 0, stream>>>(Qh, Kh, Vt, lw, Ah);
  proj_gemm_kernel<<<dim3(BN / 128, DIMC / 64), dim3(256), 0, stream>>>(Ah, WprojT, bproj, out);
}
